// AttentionModule_68848325755499
// MI455X (gfx1250) — matrix operand order repaired and checked by reading, not compiled, not run on hardware
//
#include <hip/hip_runtime.h>
#include <math.h>

// ---------------------------------------------------------------------------
// CDNA5 (gfx1250) Longformer-style encoder stack.
//  - All heavy GEMMs + attention matmuls: v_wmma_f32_16x16x32_bf16.
//  - Weights pre-converted once per layer to transposed bf16 [N][K]; all
//    GEMM producers emit bf16 shadows, so GEMM tiles are raw bf16 copies.
//  - GEMM tile staging via Tensor Data Mover (tensor_load_to_lds), LDS pad
//    matching the WMMA fragment layout, double buffered + s_wait_tensorcnt.
//  - Accumulators are named register unions (no arrays) to guarantee they
//    stay in VGPRs (no scratch spills around the WMMAs).
// wave32; 256-thread blocks = 8 waves.
// ---------------------------------------------------------------------------

typedef __attribute__((ext_vector_type(16))) __bf16 v16bf;
typedef __attribute__((ext_vector_type(8)))  float  v8f;
typedef unsigned int v4u __attribute__((ext_vector_type(4)));
typedef unsigned int v8u __attribute__((ext_vector_type(8)));

union FragB { v16bf v; uint4 q[2]; unsigned short h[16]; };
union FragC { v8f  v; float f[8]; };

__device__ __forceinline__ unsigned short f2bf(float x) {
  unsigned int u = __float_as_uint(x);
  u += 0x7fffu + ((u >> 16) & 1u);
  return (unsigned short)(u >> 16);
}
__device__ __forceinline__ float bf2f(unsigned short h) {
  return __uint_as_float(((unsigned int)h) << 16);
}

#define WMMA_BF16(A, Bv, C) \
  __builtin_amdgcn_wmma_f32_16x16x32_bf16(false, (A), false, (Bv), (short)0, (C), false, false)

#define ZERO8(CC) do { _Pragma("unroll") \
  for (int r_ = 0; r_ < 8; r_++) (CC).f[r_] = 0.f; } while (0)

__device__ __forceinline__ float wave_sum32(float v) {
  for (int m = 16; m >= 1; m >>= 1) v += __shfl_xor(v, m, 32);
  return v;
}
__device__ __forceinline__ float wave_max32(float v) {
  for (int m = 16; m >= 1; m >>= 1) v = fmaxf(v, __shfl_xor(v, m, 32));
  return v;
}
__device__ __forceinline__ float block_sum256(float v, float* red) {
  v = wave_sum32(v);
  int w = threadIdx.x >> 5;
  if ((threadIdx.x & 31) == 0) red[w] = v;
  __syncthreads();
  if (threadIdx.x == 0) { float t = 0.f; for (int i = 0; i < 8; i++) t += red[i]; red[0] = t; }
  __syncthreads();
  float r = red[0];
  __syncthreads();
  return r;
}
__device__ __forceinline__ float block_max256(float v, float* red) {
  v = wave_max32(v);
  int w = threadIdx.x >> 5;
  if ((threadIdx.x & 31) == 0) red[w] = v;
  __syncthreads();
  if (threadIdx.x == 0) { float t = -3.0e38f; for (int i = 0; i < 8; i++) t = fmaxf(t, red[i]); red[0] = t; }
  __syncthreads();
  float r = red[0];
  __syncthreads();
  return r;
}

__device__ __forceinline__ float gelu_exact(float x) {
  return 0.5f * x * (1.0f + erff(x * 0.7071067811865475f));
}

// ---------------------------------------------------------------------------
// TDM 2D tile load: bf16 tile 128 rows x 32 elems (64B rows), LDS padding
// 16B per row -> LDS row stride 80B = 40 ushorts (fragment layout below).
// ---------------------------------------------------------------------------
__device__ __forceinline__ void tdm_load_tile(
    unsigned lds_off, unsigned long long gaddr,
    unsigned tensor_d0 /*K elems*/, unsigned tensor_d1 /*rows*/,
    unsigned long long stride0 /*K elems*/)
{
  v4u g0; v8u g1;
  g0[0] = 1u;                                       // count=1, user descriptor
  g0[1] = lds_off;                                  // LDS byte address
  g0[2] = (unsigned)(gaddr & 0xffffffffu);          // global addr [95:64]
  g0[3] = (unsigned)((gaddr >> 32) & 0x1ffffffu)    // global addr [120:96]
        | (2u << 30);                               // type = 2 ("image")
  g1[0] = (1u << 16)    // data_size = 2 bytes (bf16)
        | (1u << 20)    // pad_enable
        | (3u << 22)    // pad_interval: 16 DWORDs (=64B)
        | (3u << 25);   // pad_amount: 4 DWORDs (=16B)
  g1[1] = (tensor_d0 & 0xffffu) << 16;              // tensor_dim0[15:0]
  g1[2] = (tensor_d0 >> 16) | ((tensor_d1 & 0xffffu) << 16);
  g1[3] = (tensor_d1 >> 16) | (32u << 16);          // tile_dim0 = 32 elems
  g1[4] = 128u;                                     // tile_dim1 = 128 rows
  g1[5] = (unsigned)(stride0 & 0xffffffffu);        // tensor_dim0_stride
  g1[6] = (unsigned)((stride0 >> 32) & 0xffffu);
  g1[7] = 0u;
  asm volatile("tensor_load_to_lds %0, %1" :: "s"(g0), "s"(g1));
}

// ---------------------------------------------------------------------------
// Dense GEMM: Y[M,N] = act( A[M,K](bf16) @ Bt[N,K](bf16)^T + bias[N] )
// block tile 128x128, wave tile 64x32 (4x2 WMMA), K step 32, TDM staged,
// double-buffered LDS. act: 0=none, 1=exact GELU. Outputs fp32/bf16.
// Requires K % 32 == 0; M,N % 128 == 0.
// ---------------------------------------------------------------------------
__global__ __launch_bounds__(256) void gemm_kernel(
    const unsigned short* __restrict__ A, const unsigned short* __restrict__ Bt,
    const float* __restrict__ bias, float* __restrict__ Yf,
    unsigned short* __restrict__ Ybf, int M, int K, int N, int act)
{
  __shared__ alignas(16) unsigned short As[2][128][40];
  __shared__ alignas(16) unsigned short Bs[2][128][40];

  const int tid  = threadIdx.x;
  const int lane = tid & 31, wv = tid >> 5;
  const int hi   = lane >> 4, l15 = lane & 15;
  const int wm   = wv & 1, wn = wv >> 1;
  const int m0   = blockIdx.y * 128, n0 = blockIdx.x * 128;

  FragC c00, c01, c10, c11, c20, c21, c30, c31;
  ZERO8(c00); ZERO8(c01); ZERO8(c10); ZERO8(c11);
  ZERO8(c20); ZERO8(c21); ZERO8(c30); ZERO8(c31);

  if (wv == 0) {  // prologue: DMA first K tile pair
    tdm_load_tile((unsigned)(size_t)(void*)&As[0][0][0],
                  (unsigned long long)(size_t)(A + (size_t)m0 * K),
                  (unsigned)K, (unsigned)M, (unsigned long long)K);
    tdm_load_tile((unsigned)(size_t)(void*)&Bs[0][0][0],
                  (unsigned long long)(size_t)(Bt + (size_t)n0 * K),
                  (unsigned)K, (unsigned)N, (unsigned long long)K);
  }

#define LOAD_A(AF, MI) do { \
    const unsigned short* rp_ = &As[cur][wm * 64 + (MI) * 16 + l15][0]; \
    (AF).q[0] = *(const uint4*)(rp_ + hi * 8); \
    (AF).q[1] = *(const uint4*)(rp_ + 16 + hi * 8); } while (0)
#define LOAD_B(BF, NI) do { \
    const unsigned short* rp_ = &Bs[cur][wn * 32 + (NI) * 16 + l15][0]; \
    (BF).q[0] = *(const uint4*)(rp_ + hi * 8); \
    (BF).q[1] = *(const uint4*)(rp_ + 16 + hi * 8); } while (0)

  for (int kb = 0; kb < K; kb += 32) {
    const int cur = (kb >> 5) & 1;
    if (wv == 0) {
      if (kb + 32 < K) {
        tdm_load_tile((unsigned)(size_t)(void*)&As[cur ^ 1][0][0],
                      (unsigned long long)(size_t)(A + (size_t)m0 * K + kb + 32),
                      (unsigned)K, (unsigned)M, (unsigned long long)K);
        tdm_load_tile((unsigned)(size_t)(void*)&Bs[cur ^ 1][0][0],
                      (unsigned long long)(size_t)(Bt + (size_t)n0 * K + kb + 32),
                      (unsigned)K, (unsigned)N, (unsigned long long)K);
        __builtin_amdgcn_s_wait_tensorcnt(2);  // current pair complete
      } else {
        __builtin_amdgcn_s_wait_tensorcnt(0);
      }
    }
    __syncthreads();

    FragB a0, a1, a2, a3, b0f, b1f;
    LOAD_A(a0, 0); LOAD_A(a1, 1); LOAD_A(a2, 2); LOAD_A(a3, 3);
    LOAD_B(b0f, 0); LOAD_B(b1f, 1);

    c00.v = WMMA_BF16(a0.v, b0f.v, c00.v);
    c01.v = WMMA_BF16(a0.v, b1f.v, c01.v);
    c10.v = WMMA_BF16(a1.v, b0f.v, c10.v);
    c11.v = WMMA_BF16(a1.v, b1f.v, c11.v);
    c20.v = WMMA_BF16(a2.v, b0f.v, c20.v);
    c21.v = WMMA_BF16(a2.v, b1f.v, c21.v);
    c30.v = WMMA_BF16(a3.v, b0f.v, c30.v);
    c31.v = WMMA_BF16(a3.v, b1f.v, c31.v);
    __syncthreads();
  }
#undef LOAD_A
#undef LOAD_B

#define STORE_TILE(CC, MI, NI) do { _Pragma("unroll") \
  for (int r = 0; r < 8; r++) { \
    int row = m0 + wm * 64 + (MI) * 16 + r + hi * 8; \
    int col = n0 + wn * 32 + (NI) * 16 + l15; \
    float vv = (CC).f[r] + bias[col]; \
    if (act == 1) vv = gelu_exact(vv); \
    if (Yf)  Yf[(size_t)row * N + col] = vv; \
    if (Ybf) Ybf[(size_t)row * N + col] = f2bf(vv); \
  } } while (0)

  STORE_TILE(c00, 0, 0); STORE_TILE(c01, 0, 1);
  STORE_TILE(c10, 1, 0); STORE_TILE(c11, 1, 1);
  STORE_TILE(c20, 2, 0); STORE_TILE(c21, 2, 1);
  STORE_TILE(c30, 3, 0); STORE_TILE(c31, 3, 1);
#undef STORE_TILE
}

// ---------------------------------------------------------------------------
// Weight convert + transpose: Wt[n*K + k] = bf16(W[k*N + n])
// ---------------------------------------------------------------------------
__global__ __launch_bounds__(256) void transpose_bf16_kernel(
    const float* __restrict__ W, unsigned short* __restrict__ Wt, int K, int N)
{
  __shared__ float tile[32][33];
  int n0 = blockIdx.x * 32, k0 = blockIdx.y * 32;
  int tx = threadIdx.x & 31, ty = threadIdx.x >> 5;
  for (int j = 0; j < 32; j += 8) {
    int kk = k0 + ty + j, nn = n0 + tx;
    tile[ty + j][tx] = (kk < K && nn < N) ? W[(size_t)kk * N + nn] : 0.f;
  }
  __syncthreads();
  for (int j = 0; j < 32; j += 8) {
    int nn = n0 + ty + j, kk = k0 + tx;
    if (nn < N && kk < K) Wt[(size_t)nn * K + kk] = f2bf(tile[tx][ty + j]);
  }
}

// ---------------------------------------------------------------------------
// Local (sliding-window + 3 global columns) attention over bf16 q/k/v.
// One block per (window, head, batch); 8 waves x 16 query rows; flash-style
// online softmax; WMMA for QK^T and P*V; d=64, window=128. bf16 output.
// ---------------------------------------------------------------------------
__global__ __launch_bounds__(256) void local_attn_kernel(
    const unsigned short* __restrict__ q, const unsigned short* __restrict__ k,
    const unsigned short* __restrict__ v, unsigned short* __restrict__ out,
    int S, int HID, int g0, int g1, int g2, float scale)
{
  const int blk = blockIdx.x, head = blockIdx.y, b = blockIdx.z;
  const int tid = threadIdx.x, lane = tid & 31, wv = tid >> 5;
  const int hi = lane >> 4, l15 = lane & 15;
  const int d0 = head * 64;
  const size_t base = (size_t)b * S;

  __shared__ alignas(16) unsigned short pbuf[8][16][32];

  FragB aq0, aq1;
  {
    const int qrow = blk * 128 + wv * 16 + l15;
    const unsigned short* qp = q + (base + qrow) * HID + d0;
    aq0.q[0] = *(const uint4*)(qp + hi * 8);
    aq0.q[1] = *(const uint4*)(qp + 16 + hi * 8);
    aq1.q[0] = *(const uint4*)(qp + 32 + hi * 8);
    aq1.q[1] = *(const uint4*)(qp + 32 + 16 + hi * 8);
  }

  const int gg0 = g0, gg1 = g1, gg2 = g2;
  float mrow[8], lrow[8], alpha[8];
  FragC accO0, accO1, accO2, accO3;

  // ---- init running softmax with the 3 global columns (fp32) ----
#pragma unroll
  for (int r = 0; r < 8; r++) {
    int srow = blk * 128 + wv * 16 + r + hi * 8;
    const unsigned short* qr = q + (base + srow) * HID + d0;
    const unsigned short* k0p = k + (base + gg0) * HID + d0;
    const unsigned short* k1p = k + (base + gg1) * HID + d0;
    const unsigned short* k2p = k + (base + gg2) * HID + d0;
    float s0g = 0.f, s1g = 0.f, s2g = 0.f;
    for (int dd = 0; dd < 64; dd++) {
      float qv = bf2f(qr[dd]);
      s0g += qv * bf2f(k0p[dd]);
      s1g += qv * bf2f(k1p[dd]);
      s2g += qv * bf2f(k2p[dd]);
    }
    s0g *= scale; s1g *= scale; s2g *= scale;
    float m = fmaxf(s0g, fmaxf(s1g, s2g));
    float p0g = __expf(s0g - m), p1g = __expf(s1g - m), p2g = __expf(s2g - m);
    mrow[r] = m; lrow[r] = p0g + p1g + p2g;
    const unsigned short* v0p = v + (base + gg0) * HID + d0 + l15;
    const unsigned short* v1p = v + (base + gg1) * HID + d0 + l15;
    const unsigned short* v2p = v + (base + gg2) * HID + d0 + l15;
    accO0.f[r] = p0g * bf2f(v0p[0])  + p1g * bf2f(v1p[0])  + p2g * bf2f(v2p[0]);
    accO1.f[r] = p0g * bf2f(v0p[16]) + p1g * bf2f(v1p[16]) + p2g * bf2f(v2p[16]);
    accO2.f[r] = p0g * bf2f(v0p[32]) + p1g * bf2f(v1p[32]) + p2g * bf2f(v2p[32]);
    accO3.f[r] = p0g * bf2f(v0p[48]) + p1g * bf2f(v1p[48]) + p2g * bf2f(v2p[48]);
  }

#define PV_TILE(ACC, T) do { \
    FragB bv_; \
    int dd_ = d0 + (T) * 16 + l15; \
    _Pragma("unroll") for (int e = 0; e < 16; e++) { \
      int j_ = j0 + hi * 8 + (e & 7) + (e >> 3) * 16; \
      int ak_ = blk * 128 + j_ - 128; \
      bv_.h[e] = (ak_ >= 0 && ak_ < S) ? v[(base + ak_) * HID + dd_] : (unsigned short)0; \
    } \
    (ACC).v = WMMA_BF16(pa.v, bv_.v, (ACC).v); } while (0)
#define RESCALE(ACC) do { _Pragma("unroll") \
    for (int r = 0; r < 8; r++) (ACC).f[r] *= alpha[r]; } while (0)

  for (int it = 0; it < 12; ++it) {
    const int j0 = it * 32;
    FragC s0, s1;
    ZERO8(s0); ZERO8(s1);

#pragma unroll
    for (int ks = 0; ks < 2; ++ks) {
      const v16bf aqv = (ks == 0) ? aq0.v : aq1.v;
#pragma unroll
      for (int t2 = 0; t2 < 2; ++t2) {
        FragB bk;
        int j = j0 + t2 * 16 + l15;
        int ak = blk * 128 + j - 128;
        if (ak >= 0 && ak < S) {
          const unsigned short* kp = k + (base + ak) * HID + d0 + ks * 32 + hi * 8;
          bk.q[0] = *(const uint4*)kp;
          bk.q[1] = *(const uint4*)(kp + 16);
        } else {
          _Pragma("unroll") for (int e = 0; e < 16; e++) bk.h[e] = 0;
        }
        if (t2 == 0) s0.v = WMMA_BF16(aqv, bk.v, s0.v);
        else         s1.v = WMMA_BF16(aqv, bk.v, s1.v);
      }
    }

#pragma unroll
    for (int r = 0; r < 8; r++) {
      int i = wv * 16 + r + hi * 8;
      {
        int j = j0 + l15, ak = blk * 128 + j - 128;
        int dj = j - 128 - i; if (dj < 0) dj = -dj;
        bool ok = (dj <= 128) && ak >= 0 && ak < S && ak != g0 && ak != g1 && ak != g2;
        s0.f[r] = ok ? s0.f[r] * scale : -1.0e9f;
      }
      {
        int j = j0 + 16 + l15, ak = blk * 128 + j - 128;
        int dj = j - 128 - i; if (dj < 0) dj = -dj;
        bool ok = (dj <= 128) && ak >= 0 && ak < S && ak != g0 && ak != g1 && ak != g2;
        s1.f[r] = ok ? s1.f[r] * scale : -1.0e9f;
      }
    }

    float p0a[8], p1a[8];
#pragma unroll
    for (int r = 0; r < 8; r++) {
      float m = fmaxf(s0.f[r], s1.f[r]);
      for (int msk = 8; msk >= 1; msk >>= 1) m = fmaxf(m, __shfl_xor(m, msk, 32));
      float mn = fmaxf(mrow[r], m);
      float al = __expf(mrow[r] - mn);
      float p0 = __expf(s0.f[r] - mn);
      float p1 = __expf(s1.f[r] - mn);
      float rs = p0 + p1;
      for (int msk = 8; msk >= 1; msk >>= 1) rs += __shfl_xor(rs, msk, 32);
      lrow[r] = lrow[r] * al + rs;
      mrow[r] = mn;
      alpha[r] = al; p0a[r] = p0; p1a[r] = p1;
    }
    RESCALE(accO0); RESCALE(accO1); RESCALE(accO2); RESCALE(accO3);

#pragma unroll
    for (int r = 0; r < 8; r++) {
      int rr = r + hi * 8;
      pbuf[wv][rr][l15]      = f2bf(p0a[r]);
      pbuf[wv][rr][16 + l15] = f2bf(p1a[r]);
    }
    __syncthreads();
    FragB pa;
    pa.q[0] = *(const uint4*)&pbuf[wv][l15][hi * 8];
    pa.q[1] = *(const uint4*)&pbuf[wv][l15][16 + hi * 8];

    PV_TILE(accO0, 0); PV_TILE(accO1, 1); PV_TILE(accO2, 2); PV_TILE(accO3, 3);
    __syncthreads();
  }
#undef PV_TILE
#undef RESCALE

#define OUT_TILE(ACC, T) do { _Pragma("unroll") \
  for (int r = 0; r < 8; r++) { \
    int srow_ = blk * 128 + wv * 16 + r + hi * 8; \
    out[(base + srow_) * HID + d0 + (T) * 16 + l15] = f2bf((ACC).f[r] / lrow[r]); \
  } } while (0)

  OUT_TILE(accO0, 0); OUT_TILE(accO1, 1); OUT_TILE(accO2, 2); OUT_TILE(accO3, 3);
#undef OUT_TILE
}

// ---------------------------------------------------------------------------
// qg rows: out[b][g][:] = h[g_row] @ Wqg + bqg   (fp32; only 3 rows needed)
// ---------------------------------------------------------------------------
__global__ __launch_bounds__(256) void rowgemm_kernel(
    const float* __restrict__ h, const float* __restrict__ W,
    const float* __restrict__ bias, float* __restrict__ out,
    int S, int HID, int g0, int g1, int g2)
{
  int g = blockIdx.x, b = blockIdx.y;
  int gr = (g == 0) ? g0 : ((g == 1) ? g1 : g2);
  const float* hr = h + ((size_t)b * S + gr) * HID;
  for (int n = threadIdx.x; n < HID; n += 256) {
    float a = bias[n];
    for (int kk = 0; kk < HID; kk++) a += hr[kk] * W[(size_t)kk * HID + n];
    out[((size_t)b * 3 + g) * HID + n] = a;
  }
}

// ---------------------------------------------------------------------------
// Full attention for the 3 global rows (overrides local output, bf16 k/v).
// grid (3, NH, B); dyn-shared: (S + 16) floats.
// ---------------------------------------------------------------------------
__global__ __launch_bounds__(256) void global_attn_kernel(
    const float* __restrict__ qg, const unsigned short* __restrict__ kgf,
    const unsigned short* __restrict__ vgf, unsigned short* __restrict__ out,
    int S, int HID, int g0, int g1, int g2, float scale)
{
  extern __shared__ float smem[];
  float* sc  = smem;
  float* red = smem + S;
  int g = blockIdx.x, head = blockIdx.y, b = blockIdx.z;
  int grow = (g == 0) ? g0 : ((g == 1) ? g1 : g2);
  const float* qr = qg + ((size_t)b * 3 + g) * HID + head * 64;

  for (int s = threadIdx.x; s < S; s += 256) {
    const unsigned short* kr = kgf + ((size_t)b * S + s) * HID + head * 64;
    float a = 0.f;
    for (int dd = 0; dd < 64; dd++) a += qr[dd] * bf2f(kr[dd]);
    sc[s] = a * scale;
  }
  __syncthreads();

  float lm = -3.0e38f;
  for (int s = threadIdx.x; s < S; s += 256) lm = fmaxf(lm, sc[s]);
  float m = block_max256(lm, red);
  float ls = 0.f;
  for (int s = threadIdx.x; s < S; s += 256) ls += __expf(sc[s] - m);
  float l = block_sum256(ls, red);

  if (threadIdx.x < 64) {
    int dd = threadIdx.x;
    float a = 0.f;
    for (int s = 0; s < S; s++)
      a += __expf(sc[s] - m) * bf2f(vgf[((size_t)b * S + s) * HID + head * 64 + dd]);
    out[((size_t)b * S + grow) * HID + head * 64 + dd] = f2bf(a / l);
  }
}

// ---------------------------------------------------------------------------
// Embedding assembly + LayerNorm; writes fp32 master + bf16 shadow.
// ---------------------------------------------------------------------------
__global__ __launch_bounds__(256) void embed_ln_kernel(
    const float* __restrict__ x0, const float* __restrict__ x1,
    const float* __restrict__ cls, const float* __restrict__ sep,
    const float* __restrict__ pos, const float* __restrict__ te,
    const float* __restrict__ lnw, const float* __restrict__ lnb,
    float* __restrict__ out, unsigned short* __restrict__ outbf,
    int S0, int S1, int S, int HID)
{
  __shared__ float red[8];
  int s = blockIdx.x, b = blockIdx.y;
  const float* src;
  if (s == 0)            src = cls;
  else if (s <= S0)      src = x0 + ((size_t)b * S0 + (s - 1)) * HID;
  else if (s == S0 + 1)  src = sep;
  else if (s < S - 1)    src = x1 + ((size_t)b * S1 + (s - S0 - 2)) * HID;
  else                   src = sep;
  int tt = (s >= S0 + 2) ? 1 : 0;

  float cache[4];
  int cnt = 0; float sum = 0.f;
  for (int c = threadIdx.x; c < HID; c += 256) {
    float vv = src[c] + pos[(size_t)(s + 2) * HID + c] + te[(size_t)tt * HID + c];
    cache[cnt++] = vv; sum += vv;
  }
  float mu = block_sum256(sum, red) / (float)HID;
  float sq = 0.f;
  for (int i = 0; i < cnt; i++) { float d = cache[i] - mu; sq += d * d; }
  float var = block_sum256(sq, red) / (float)HID;
  float rstd = rsqrtf(var + 1e-5f);
  cnt = 0;
  for (int c = threadIdx.x; c < HID; c += 256) {
    float y = (cache[cnt++] - mu) * rstd * lnw[c] + lnb[c];
    size_t idx = ((size_t)b * S + s) * HID + c;
    out[idx] = y; outbf[idx] = f2bf(y);
  }
}

// ---------------------------------------------------------------------------
// h = LayerNorm(h + t); writes fp32 master + bf16 shadow.
// ---------------------------------------------------------------------------
__global__ __launch_bounds__(256) void add_ln_kernel(
    const float* __restrict__ a, const float* __restrict__ t,
    const float* __restrict__ lnw, const float* __restrict__ lnb,
    float* __restrict__ out, unsigned short* __restrict__ outbf,
    int S, int HID)
{
  __shared__ float red[8];
  size_t row = (size_t)blockIdx.y * S + blockIdx.x;
  float cache[4];
  int cnt = 0; float sum = 0.f;
  for (int c = threadIdx.x; c < HID; c += 256) {
    float vv = a[row * HID + c] + t[row * HID + c];
    cache[cnt++] = vv; sum += vv;
  }
  float mu = block_sum256(sum, red) / (float)HID;
  float sq = 0.f;
  for (int i = 0; i < cnt; i++) { float d = cache[i] - mu; sq += d * d; }
  float var = block_sum256(sq, red) / (float)HID;
  float rstd = rsqrtf(var + 1e-5f);
  cnt = 0;
  for (int c = threadIdx.x; c < HID; c += 256) {
    float y = (cache[cnt++] - mu) * rstd * lnw[c] + lnb[c];
    out[row * HID + c] = y; outbf[row * HID + c] = f2bf(y);
  }
}

// ---------------------------------------------------------------------------
// pooled = tanh(h[:,0] @ Wp + bp)   (fp32 output)
// ---------------------------------------------------------------------------
__global__ __launch_bounds__(256) void pooled_kernel(
    const float* __restrict__ h, const float* __restrict__ Wp,
    const float* __restrict__ bp, float* __restrict__ out, int S, int HID)
{
  int b = blockIdx.x;
  const float* hr = h + (size_t)b * S * HID;
  for (int n = threadIdx.x; n < HID; n += 256) {
    float a = bp[n];
    for (int kk = 0; kk < HID; kk++) a += hr[kk] * Wp[(size_t)kk * HID + n];
    out[(size_t)b * HID + n] = tanhf(a);
  }
}

// ---------------------------------------------------------------------------
extern "C" void kernel_launch(void* const* d_in, const int* in_sizes, int n_in,
                              void* d_out, int out_size, void* d_ws, size_t ws_size,
                              hipStream_t stream)
{
  const float* x0   = (const float*)d_in[0];
  const float* x1   = (const float*)d_in[1];
  const float* cls  = (const float*)d_in[2];
  const float* sep  = (const float*)d_in[3];
  const float* pos  = (const float*)d_in[4];
  const float* te   = (const float*)d_in[5];
  const float* elnw = (const float*)d_in[6];
  const float* elnb = (const float*)d_in[7];

  const int HID = in_sizes[2];
  const int B   = out_size / HID;
  const int S0  = in_sizes[0] / (B * HID);
  const int S1  = in_sizes[1] / (B * HID);
  const int L   = in_sizes[8] / (HID * HID);
  const int FF  = in_sizes[25] / L;
  const int S   = S0 + S1 + 3;                 // 4096, multiple of 128
  const int nb  = S / 128;
  const int g0 = 0, g1 = 1 + S0, g2 = S - 1;
  const int D  = 64;                           // head dim
  const int NH = HID / D;
  const float scale = 1.0f / sqrtf((float)D);
  const int M = B * S;

  const size_t SLOT = (size_t)M * HID;
  char* w = (char*)d_ws;
  float* hbuf = (float*)w;            w += SLOT * 4;
  float* tmpb = (float*)w;            w += SLOT * 4;
  float* qgb  = (float*)w;            w += (size_t)B * 3 * HID * 4;
  unsigned short* hbf   = (unsigned short*)w;  w += SLOT * 2;
  unsigned short* qbf   = (unsigned short*)w;  w += SLOT * 2;
  unsigned short* kbf   = (unsigned short*)w;  w += SLOT * 2;
  unsigned short* vbf   = (unsigned short*)w;  w += SLOT * 2;
  unsigned short* kgfbf = (unsigned short*)w;  w += SLOT * 2;
  unsigned short* vgfbf = (unsigned short*)w;  w += SLOT * 2;
  unsigned short* attnbf= (unsigned short*)w;  w += SLOT * 2;
  unsigned short* ffibf = (unsigned short*)w;  w += (size_t)M * FF * 2;
  unsigned short* WT    = (unsigned short*)w;  w += (size_t)FF * HID * 2;

  dim3 blk256(256);
  embed_ln_kernel<<<dim3(S, B), blk256, 0, stream>>>(
      x0, x1, cls, sep, pos, te, elnw, elnb, hbuf, hbf, S0, S1, S, HID);

  dim3 gT((HID + 31) / 32, (HID + 31) / 32);
  dim3 gTi((FF + 31) / 32, (HID + 31) / 32);
  dim3 gTf((HID + 31) / 32, (FF + 31) / 32);
  dim3 gN(HID / 128, M / 128);
  dim3 gF(FF / 128, M / 128);

  for (int l = 0; l < L; ++l) {
    const float* Wq  = (const float*)d_in[8]  + (size_t)l * HID * HID;
    const float* bq  = (const float*)d_in[9]  + (size_t)l * HID;
    const float* Wk  = (const float*)d_in[10] + (size_t)l * HID * HID;
    const float* bk_ = (const float*)d_in[11] + (size_t)l * HID;
    const float* Wv  = (const float*)d_in[12] + (size_t)l * HID * HID;
    const float* bv_ = (const float*)d_in[13] + (size_t)l * HID;
    const float* Wqg = (const float*)d_in[14] + (size_t)l * HID * HID;
    const float* bqg = (const float*)d_in[15] + (size_t)l * HID;
    const float* Wkg = (const float*)d_in[16] + (size_t)l * HID * HID;
    const float* bkg = (const float*)d_in[17] + (size_t)l * HID;
    const float* Wvg = (const float*)d_in[18] + (size_t)l * HID * HID;
    const float* bvg = (const float*)d_in[19] + (size_t)l * HID;
    const float* Wo  = (const float*)d_in[20] + (size_t)l * HID * HID;
    const float* bo  = (const float*)d_in[21] + (size_t)l * HID;
    const float* l1w = (const float*)d_in[22] + (size_t)l * HID;
    const float* l1b = (const float*)d_in[23] + (size_t)l * HID;
    const float* Wi  = (const float*)d_in[24] + (size_t)l * HID * FF;
    const float* bi  = (const float*)d_in[25] + (size_t)l * FF;
    const float* Wff = (const float*)d_in[26] + (size_t)l * FF * HID;
    const float* bff = (const float*)d_in[27] + (size_t)l * HID;
    const float* l2w = (const float*)d_in[28] + (size_t)l * HID;
    const float* l2b = (const float*)d_in[29] + (size_t)l * HID;

    transpose_bf16_kernel<<<gT, blk256, 0, stream>>>(Wq, WT, HID, HID);
    gemm_kernel<<<gN, blk256, 0, stream>>>(hbf, WT, bq,  nullptr, qbf,   M, HID, HID, 0);
    transpose_bf16_kernel<<<gT, blk256, 0, stream>>>(Wk, WT, HID, HID);
    gemm_kernel<<<gN, blk256, 0, stream>>>(hbf, WT, bk_, nullptr, kbf,   M, HID, HID, 0);
    transpose_bf16_kernel<<<gT, blk256, 0, stream>>>(Wv, WT, HID, HID);
    gemm_kernel<<<gN, blk256, 0, stream>>>(hbf, WT, bv_, nullptr, vbf,   M, HID, HID, 0);
    transpose_bf16_kernel<<<gT, blk256, 0, stream>>>(Wkg, WT, HID, HID);
    gemm_kernel<<<gN, blk256, 0, stream>>>(hbf, WT, bkg, nullptr, kgfbf, M, HID, HID, 0);
    transpose_bf16_kernel<<<gT, blk256, 0, stream>>>(Wvg, WT, HID, HID);
    gemm_kernel<<<gN, blk256, 0, stream>>>(hbf, WT, bvg, nullptr, vgfbf, M, HID, HID, 0);
    rowgemm_kernel<<<dim3(3, B), blk256, 0, stream>>>(hbuf, Wqg, bqg, qgb, S, HID, g0, g1, g2);

    local_attn_kernel<<<dim3(nb, NH, B), blk256, 0, stream>>>(
        qbf, kbf, vbf, attnbf, S, HID, g0, g1, g2, scale);
    global_attn_kernel<<<dim3(3, NH, B), blk256, (S + 16) * sizeof(float), stream>>>(
        qgb, kgfbf, vgfbf, attnbf, S, HID, g0, g1, g2, scale);

    transpose_bf16_kernel<<<gT, blk256, 0, stream>>>(Wo, WT, HID, HID);
    gemm_kernel<<<gN, blk256, 0, stream>>>(attnbf, WT, bo, tmpb, nullptr, M, HID, HID, 0);
    add_ln_kernel<<<dim3(S, B), blk256, 0, stream>>>(hbuf, tmpb, l1w, l1b, hbuf, hbf, S, HID);

    transpose_bf16_kernel<<<gTi, blk256, 0, stream>>>(Wi, WT, HID, FF);
    gemm_kernel<<<gF, blk256, 0, stream>>>(hbf, WT, bi, nullptr, ffibf, M, HID, FF, 1);
    transpose_bf16_kernel<<<gTf, blk256, 0, stream>>>(Wff, WT, FF, HID);
    gemm_kernel<<<gN, blk256, 0, stream>>>(ffibf, WT, bff, tmpb, nullptr, M, FF, HID, 0);
    add_ln_kernel<<<dim3(S, B), blk256, 0, stream>>>(hbuf, tmpb, l2w, l2b, hbuf, hbf, S, HID);
  }

  pooled_kernel<<<dim3(B), blk256, 0, stream>>>(
      hbuf, (const float*)d_in[30], (const float*)d_in[31], (float*)d_out, S, HID);
}
